// Shared_bottom_87282325389516
// MI455X (gfx1250) — matrix operand order repaired and checked by reading, not compiled, not run on hardware
//
#include <hip/hip_runtime.h>
#include <hip/hip_bf16.h>

// ---------------- types ----------------
typedef __bf16 bf16;
typedef __attribute__((ext_vector_type(16))) __bf16 v16bf;
typedef __attribute__((ext_vector_type(8)))  __bf16 v8bf;
typedef __attribute__((ext_vector_type(8)))  float  v8f;

// ---------------- problem dims ----------------
#define IN_SZ   512
#define H1_SZ   1024
#define H2_SZ   512
#define OUT_SZ  64
#define NEXP    8
#define BATCH   8192
#define TILE_M  32                  // routing tile height (2 WMMA row-blocks)
#define MAXTILES 264                // 8192/32 + 8 padding tiles
#define MAXROWS (MAXTILES * TILE_M) // 8448

// ---------------- workspace layout (bytes) ----------------
#define OFF_WINT   ((size_t)0)                               // 1024*512 bf16
#define OFF_INA    (OFF_WINT  + (size_t)H1_SZ*IN_SZ*2)       // 8192*512 bf16
#define OFF_MID    (OFF_INA   + (size_t)BATCH*IN_SZ*2)       // 8192*1024 bf16
#define OFF_W1T    (OFF_MID   + (size_t)BATCH*H1_SZ*2)       // 8*512*1024 bf16
#define OFF_W2T    (OFF_W1T   + (size_t)NEXP*H2_SZ*H1_SZ*2)  // 8*64*512 bf16
#define OFF_H      (OFF_W2T   + (size_t)NEXP*OUT_SZ*H2_SZ*2) // 8448*512 bf16
#define OFF_PERM   (OFF_H     + (size_t)MAXROWS*H2_SZ*2)
#define OFF_TILEE  (OFF_PERM  + (size_t)MAXROWS*4)
#define OFF_CTR    (OFF_TILEE + (size_t)MAXTILES*4)          // fill[8], nTiles[1]

// ---------------- device helpers ----------------
__device__ __forceinline__ float eluf(float x) {
    return x > 0.0f ? x : (__expf(x) - 1.0f);
}

// A fragment (16x32 bf16): elems 0..15 = K=2*kb..2*kb+15 (contiguous, matches B),
// kb = (lane>>4)*8. Two aligned 16B loads per lane.
__device__ __forceinline__ v16bf load_a_frag(const bf16* __restrict__ rowk0, int kb) {
    v8bf lo = *(const v8bf*)(rowk0 + 2 * kb);
    v8bf hi = *(const v8bf*)(rowk0 + 2 * kb + 8);
    v16bf r;
#pragma unroll
    for (int j = 0; j < 8; ++j) { r[j] = lo[j]; r[j + 8] = hi[j]; }
    return r;
}

__device__ __forceinline__ v8f wmma_bf16(v16bf a, v16bf b, v8f c) {
    return __builtin_amdgcn_wmma_f32_16x16x32_bf16(
        /*neg_a=*/false, a, /*neg_b=*/false, b,
        /*c_mod=*/(short)0, c, /*reuse_a=*/false, /*reuse_b=*/false);
}

// ---------------- conversion kernels ----------------
__global__ __launch_bounds__(256) void k_cvt_f32_bf16(const float* __restrict__ src,
                                                      bf16* __restrict__ dst, int n) {
    int stride = gridDim.x * blockDim.x;
    for (int i = blockIdx.x * blockDim.x + threadIdx.x; i < n; i += stride)
        dst[i] = (bf16)src[i];
}

// W_in [512,1024] -> WinT [n=1024][k=512] bf16
__global__ __launch_bounds__(256) void k_cvt_WinT(const float* __restrict__ W,
                                                  bf16* __restrict__ WT) {
    int stride = gridDim.x * blockDim.x;
    for (int i = blockIdx.x * blockDim.x + threadIdx.x; i < H1_SZ * IN_SZ; i += stride) {
        int n = i >> 9, k = i & (IN_SZ - 1);
        WT[i] = (bf16)W[(size_t)k * H1_SZ + n];
    }
}

// W1 [8,1024,512] -> W1T [e][n=512][k=1024] bf16
__global__ __launch_bounds__(256) void k_cvt_W1T(const float* __restrict__ W,
                                                 bf16* __restrict__ WT) {
    int stride = gridDim.x * blockDim.x;
    const int per = H2_SZ * H1_SZ;
    for (int i = blockIdx.x * blockDim.x + threadIdx.x; i < NEXP * per; i += stride) {
        int e = i / per, rem = i - e * per;
        int n = rem >> 10, k = rem & (H1_SZ - 1);
        WT[i] = (bf16)W[(size_t)e * per + (size_t)k * H2_SZ + n];
    }
}

// W2 [8,512,64] -> W2T [e][o=64][k=512] bf16
__global__ __launch_bounds__(256) void k_cvt_W2T(const float* __restrict__ W,
                                                 bf16* __restrict__ WT) {
    int stride = gridDim.x * blockDim.x;
    const int per = OUT_SZ * H2_SZ;
    for (int i = blockIdx.x * blockDim.x + threadIdx.x; i < NEXP * per; i += stride) {
        int e = i / per, rem = i - e * per;
        int o = rem >> 9, k = rem & (H2_SZ - 1);
        WT[i] = (bf16)W[(size_t)e * per + (size_t)k * OUT_SZ + o];
    }
}

// ---------------- shared-bottom GEMM: mid = ELU(inA @ W_in + b_in) ----------------
// grid (128, 4), block 256 (8 waves = 2M x 4N). Wave tile: M=32, N=64. K=512.
// Register double-buffered K-loop: loads for k+1 issued before WMMAs of k.
__global__ __launch_bounds__(256) void k_gemm_bottom(const bf16* __restrict__ A,
                                                     const bf16* __restrict__ WT,
                                                     const float* __restrict__ bias,
                                                     bf16* __restrict__ mid) {
    const int lane = threadIdx.x & 31, wid = threadIdx.x >> 5;
    const int laneLo = lane & 15, half = lane >> 4;
    const int waveM = wid >> 2, waveNq = wid & 3;
    const int Mbase = blockIdx.x * 64 + waveM * 32;
    const int waveN = blockIdx.y * 256 + waveNq * 64;
    const int kb = half * 8;

    const bf16* arow0 = A + (size_t)(Mbase + laneLo) * IN_SZ;
    const bf16* arow1 = arow0 + (size_t)16 * IN_SZ;
    const bf16* bcol[4];
#pragma unroll
    for (int nt = 0; nt < 4; ++nt)
        bcol[nt] = WT + (size_t)(waveN + nt * 16 + laneLo) * IN_SZ + half * 16;

    v8f acc[2][4];
#pragma unroll
    for (int m = 0; m < 2; ++m)
#pragma unroll
        for (int nt = 0; nt < 4; ++nt) acc[m][nt] = (v8f)0.0f;

    v16bf a0 = load_a_frag(arow0, kb);
    v16bf a1 = load_a_frag(arow1, kb);
    v16bf b[4];
#pragma unroll
    for (int nt = 0; nt < 4; ++nt) b[nt] = *(const v16bf*)(bcol[nt]);

#pragma unroll 2
    for (int k0 = 32; k0 < IN_SZ; k0 += 32) {
        v16bf a0n = load_a_frag(arow0 + k0, kb);
        v16bf a1n = load_a_frag(arow1 + k0, kb);
        v16bf bn[4];
#pragma unroll
        for (int nt = 0; nt < 4; ++nt) bn[nt] = *(const v16bf*)(bcol[nt] + k0);
        __builtin_prefetch(arow0 + k0 + 32, 0, 1);
#pragma unroll
        for (int nt = 0; nt < 4; ++nt) {
            acc[0][nt] = wmma_bf16(a0, b[nt], acc[0][nt]);
            acc[1][nt] = wmma_bf16(a1, b[nt], acc[1][nt]);
        }
        a0 = a0n; a1 = a1n;
#pragma unroll
        for (int nt = 0; nt < 4; ++nt) b[nt] = bn[nt];
    }
#pragma unroll
    for (int nt = 0; nt < 4; ++nt) {
        acc[0][nt] = wmma_bf16(a0, b[nt], acc[0][nt]);
        acc[1][nt] = wmma_bf16(a1, b[nt], acc[1][nt]);
    }

#pragma unroll
    for (int nt = 0; nt < 4; ++nt) {
        int col = waveN + nt * 16 + laneLo;
        float bv = bias[col];
#pragma unroll
        for (int m = 0; m < 2; ++m)
#pragma unroll
            for (int r = 0; r < 8; ++r) {
                int row = Mbase + m * 16 + r + half * 8;
                mid[(size_t)row * H1_SZ + col] = (bf16)eluf(acc[m][nt][r] + bv);
            }
    }
}

// ---------------- routing (32-row tiles) ----------------
__global__ __launch_bounds__(256) void k_route_setup(const int* __restrict__ cat,
                                                     int* __restrict__ ctr,        // fill[8], nTiles[1]
                                                     int* __restrict__ tileExpert,
                                                     int* __restrict__ perm) {
    __shared__ int cnt[NEXP];
    __shared__ int padEnd;
    int tid = threadIdx.x;
    if (tid < NEXP) cnt[tid] = 0;
    __syncthreads();
    for (int i = tid; i < BATCH; i += 256) atomicAdd(&cnt[cat[i]], 1);
    __syncthreads();
    if (tid == 0) {
        int off = 0, tiles = 0;
        for (int e = 0; e < NEXP; ++e) {
            ctr[e] = off;                       // fill counter = group base
            int pc = (cnt[e] + (TILE_M - 1)) & ~(TILE_M - 1);
            int nt = pc / TILE_M;
            for (int j = 0; j < nt; ++j) tileExpert[tiles + j] = e;
            tiles += nt;
            off += pc;
        }
        ctr[8] = tiles;
        padEnd = off;
    }
    __syncthreads();
    int P = padEnd;
    for (int i = tid; i < P; i += 256) perm[i] = -1;
}

__global__ __launch_bounds__(256) void k_route_fill(const int* __restrict__ cat,
                                                    int* __restrict__ ctr,
                                                    int* __restrict__ perm) {
    int i = blockIdx.x * blockDim.x + threadIdx.x;
    if (i < BATCH) {
        int pos = atomicAdd(&ctr[cat[i]], 1);
        perm[pos] = i;
    }
}

// ---------------- expert layer 1: h = ELU(mid[perm] @ W1[e] + b1[e]) ----------------
// one block per 32-row tile; 8 waves x N=64 = 512; wave M=32; K=1024, double-buffered.
__global__ __launch_bounds__(256) void k_expert1(const bf16* __restrict__ mid,
                                                 const bf16* __restrict__ W1T,
                                                 const float* __restrict__ b1,
                                                 const int* __restrict__ perm,
                                                 const int* __restrict__ tileExpert,
                                                 const int* __restrict__ nTiles,
                                                 bf16* __restrict__ h) {
    const int t = blockIdx.x;
    if (t >= nTiles[0]) return;                 // block-uniform branch (EXEC stays all-1)
    const int e = tileExpert[t];

    const int lane = threadIdx.x & 31, wid = threadIdx.x >> 5;
    const int laneLo = lane & 15, half = lane >> 4;
    const int kb = half * 8;

    int ri0 = perm[t * TILE_M + laneLo];
    int ri1 = perm[t * TILE_M + 16 + laneLo];
    const bf16* arow0 = mid + (size_t)(ri0 < 0 ? 0 : ri0) * H1_SZ;
    const bf16* arow1 = mid + (size_t)(ri1 < 0 ? 0 : ri1) * H1_SZ;
    const bf16* wbase = W1T + (size_t)e * H2_SZ * H1_SZ;
    const int waveN = wid * 64;
    const bf16* bcol[4];
#pragma unroll
    for (int nt = 0; nt < 4; ++nt)
        bcol[nt] = wbase + (size_t)(waveN + nt * 16 + laneLo) * H1_SZ + half * 16;

    v8f acc[2][4];
#pragma unroll
    for (int m = 0; m < 2; ++m)
#pragma unroll
        for (int nt = 0; nt < 4; ++nt) acc[m][nt] = (v8f)0.0f;

    v16bf a0 = load_a_frag(arow0, kb);
    v16bf a1 = load_a_frag(arow1, kb);
    v16bf b[4];
#pragma unroll
    for (int nt = 0; nt < 4; ++nt) b[nt] = *(const v16bf*)(bcol[nt]);

#pragma unroll 2
    for (int k0 = 32; k0 < H1_SZ; k0 += 32) {
        v16bf a0n = load_a_frag(arow0 + k0, kb);
        v16bf a1n = load_a_frag(arow1 + k0, kb);
        v16bf bn[4];
#pragma unroll
        for (int nt = 0; nt < 4; ++nt) bn[nt] = *(const v16bf*)(bcol[nt] + k0);
        __builtin_prefetch(arow0 + k0 + 32, 0, 1);
#pragma unroll
        for (int nt = 0; nt < 4; ++nt) {
            acc[0][nt] = wmma_bf16(a0, b[nt], acc[0][nt]);
            acc[1][nt] = wmma_bf16(a1, b[nt], acc[1][nt]);
        }
        a0 = a0n; a1 = a1n;
#pragma unroll
        for (int nt = 0; nt < 4; ++nt) b[nt] = bn[nt];
    }
#pragma unroll
    for (int nt = 0; nt < 4; ++nt) {
        acc[0][nt] = wmma_bf16(a0, b[nt], acc[0][nt]);
        acc[1][nt] = wmma_bf16(a1, b[nt], acc[1][nt]);
    }

#pragma unroll
    for (int nt = 0; nt < 4; ++nt) {
        int col = waveN + nt * 16 + laneLo;
        float bv = b1[e * H2_SZ + col];
#pragma unroll
        for (int m = 0; m < 2; ++m)
#pragma unroll
            for (int r = 0; r < 8; ++r) {
                int rt = m * 16 + r + half * 8;
                h[((size_t)t * TILE_M + rt) * H2_SZ + col] = (bf16)eluf(acc[m][nt][r] + bv);
            }
    }
}

// ---------------- expert layer 2: out[row] = h @ W2[e] + b2[e], scattered ----------------
// one block per 32-row tile; 4 waves cover O=64; wave M=32,N=16; K=512, double-buffered.
__global__ __launch_bounds__(128) void k_expert2(const bf16* __restrict__ h,
                                                 const bf16* __restrict__ W2T,
                                                 const float* __restrict__ b2,
                                                 const int* __restrict__ perm,
                                                 const int* __restrict__ tileExpert,
                                                 const int* __restrict__ nTiles,
                                                 float* __restrict__ out) {
    const int t = blockIdx.x;
    if (t >= nTiles[0]) return;
    const int e = tileExpert[t];

    const int lane = threadIdx.x & 31, wid = threadIdx.x >> 5;
    const int laneLo = lane & 15, half = lane >> 4;
    const int kb = half * 8;

    const bf16* arow0 = h + ((size_t)t * TILE_M + laneLo) * H2_SZ;
    const bf16* arow1 = arow0 + (size_t)16 * H2_SZ;
    const bf16* wrow = W2T + (size_t)e * OUT_SZ * H2_SZ
                     + (size_t)(wid * 16 + laneLo) * H2_SZ + half * 16;

    v8f acc0 = (v8f)0.0f, acc1 = (v8f)0.0f;
    v16bf a0 = load_a_frag(arow0, kb);
    v16bf a1 = load_a_frag(arow1, kb);
    v16bf b = *(const v16bf*)(wrow);

#pragma unroll 2
    for (int k0 = 32; k0 < H2_SZ; k0 += 32) {
        v16bf a0n = load_a_frag(arow0 + k0, kb);
        v16bf a1n = load_a_frag(arow1 + k0, kb);
        v16bf bn  = *(const v16bf*)(wrow + k0);
        acc0 = wmma_bf16(a0, b, acc0);
        acc1 = wmma_bf16(a1, b, acc1);
        a0 = a0n; a1 = a1n; b = bn;
    }
    acc0 = wmma_bf16(a0, b, acc0);
    acc1 = wmma_bf16(a1, b, acc1);

    int col = wid * 16 + laneLo;
    float bv = b2[e * OUT_SZ + col];
#pragma unroll
    for (int r = 0; r < 8; ++r) {
        int row0 = perm[t * TILE_M + r + half * 8];
        if (row0 >= 0) out[(size_t)row0 * OUT_SZ + col] = acc0[r] + bv;
        int row1 = perm[t * TILE_M + 16 + r + half * 8];
        if (row1 >= 0) out[(size_t)row1 * OUT_SZ + col] = acc1[r] + bv;
    }
}

// ---------------- launcher ----------------
extern "C" void kernel_launch(void* const* d_in, const int* in_sizes, int n_in,
                              void* d_out, int out_size, void* d_ws, size_t ws_size,
                              hipStream_t stream) {
    const float* input    = (const float*)d_in[0];
    const int*   category = (const int*)d_in[1];
    const float* W_in     = (const float*)d_in[2];
    const float* b_in     = (const float*)d_in[3];
    const float* W1       = (const float*)d_in[4];
    const float* b1       = (const float*)d_in[5];
    const float* W2       = (const float*)d_in[6];
    const float* b2       = (const float*)d_in[7];
    float* out = (float*)d_out;

    char* ws = (char*)d_ws;
    bf16* WinT = (bf16*)(ws + OFF_WINT);
    bf16* inA  = (bf16*)(ws + OFF_INA);
    bf16* mid  = (bf16*)(ws + OFF_MID);
    bf16* W1T  = (bf16*)(ws + OFF_W1T);
    bf16* W2T  = (bf16*)(ws + OFF_W2T);
    bf16* hbuf = (bf16*)(ws + OFF_H);
    int*  perm = (int*)(ws + OFF_PERM);
    int*  tileExpert = (int*)(ws + OFF_TILEE);
    int*  ctr  = (int*)(ws + OFF_CTR);   // [0..7]=fill counters, [8]=nTiles

    // precision conversion / weight transposition
    k_cvt_f32_bf16<<<2048, 256, 0, stream>>>(input, inA, BATCH * IN_SZ);
    k_cvt_WinT    <<< 512, 256, 0, stream>>>(W_in, WinT);
    k_cvt_W1T     <<<2048, 256, 0, stream>>>(W1, W1T);
    k_cvt_W2T     <<< 256, 256, 0, stream>>>(W2, W2T);

    // shared bottom
    k_gemm_bottom<<<dim3(BATCH / 64, H1_SZ / 256), 256, 0, stream>>>(inA, WinT, b_in, mid);

    // hard routing: group rows by expert, pad groups to 32-row tiles
    k_route_setup<<<1, 256, 0, stream>>>(category, ctr, tileExpert, perm);
    k_route_fill <<<BATCH / 256, 256, 0, stream>>>(category, ctr, perm);

    // routed expert MLP
    k_expert1<<<MAXTILES, 256, 0, stream>>>(mid, W1T, b1, perm, tileExpert, ctr + 8, hbuf);
    k_expert2<<<MAXTILES, 128, 0, stream>>>(hbuf, W2T, b2, perm, tileExpert, ctr + 8, out);
}
